// WordGraphNetwork_88072599371949
// MI455X (gfx1250) — matrix operand order repaired and checked by reading, not compiled, not run on hardware
//
#include <hip/hip_runtime.h>
#include <hip/hip_bf16.h>

#define ND 128

typedef __attribute__((ext_vector_type(16))) __bf16 v16bf;
typedef __attribute__((ext_vector_type(8)))  float  v8f;

// ---------- helpers ----------

// Let clang lower fptrunc f32->bf16 to the native cvt on gfx1250 (RNE).
__device__ __forceinline__ __bf16 f2bf(float f) { return (__bf16)f; }

__device__ __forceinline__ float sigm(float x) {
    return 1.0f / (1.0f + __expf(-x));
}

__device__ __forceinline__ v8f wmma_bf16(v16bf a, v16bf b, v8f c) {
#if defined(__HIP_DEVICE_COMPILE__)
    return __builtin_amdgcn_wmma_f32_16x16x32_bf16(
        /*neg_a=*/false, a, /*neg_b=*/false, b,
        /*c_mod=*/(short)0, c, /*reuse_a=*/false, /*reuse_b=*/false);
#else
    (void)a; (void)b; return c;
#endif
}

// A fragment: 16x32 bf16, contiguous k order (same as load_bfrag):
// lane<16: row=lane, K = 0..15; lane>=16: row=lane-16, K = 16..31
// Works for global or LDS-backed rows (addrspace inferred after inlining).
__device__ __forceinline__ v16bf load_afrag(const float* Arow, int lane, int k0) {
    const int r  = lane & 15;
    const int hf = lane >> 4;
    const float* base = Arow + r * ND + k0 + hf * 16;
    const float4 x0 = *(const float4*)(base);
    const float4 x1 = *(const float4*)(base + 4);
    const float4 x2 = *(const float4*)(base + 8);
    const float4 x3 = *(const float4*)(base + 12);
    v16bf a;
    a[0]=f2bf(x0.x); a[1]=f2bf(x0.y); a[2]=f2bf(x0.z); a[3]=f2bf(x0.w);
    a[4]=f2bf(x1.x); a[5]=f2bf(x1.y); a[6]=f2bf(x1.z); a[7]=f2bf(x1.w);
    a[8]=f2bf(x2.x); a[9]=f2bf(x2.y); a[10]=f2bf(x2.z); a[11]=f2bf(x2.w);
    a[12]=f2bf(x3.x); a[13]=f2bf(x3.y); a[14]=f2bf(x3.z); a[15]=f2bf(x3.w);
    return a;
}

// B fragment: 32x16 bf16 from transposed weights Wt[n*ND + k].
__device__ __forceinline__ v16bf load_bfrag(const __bf16* __restrict__ Wt, int lane, int ntile, int k0) {
    const int n  = ntile * 16 + (lane & 15);
    const int kb = k0 + (lane >> 4) * 16;
    return *(const v16bf*)(Wt + n * ND + kb);
}

// C/D tile: VGPR v, lane -> row m = v + (lane>>4)*8, col n = ntile*16 + (lane&15)
__device__ __forceinline__ void load_ctile(const float* __restrict__ Crow, int lane, int ntile, float o[8]) {
    const int n  = ntile * 16 + (lane & 15);
    const int hf = lane >> 4;
#pragma unroll
    for (int v = 0; v < 8; ++v) o[v] = Crow[(v + hf * 8) * ND + n];
}

__device__ __forceinline__ void store_ctile(float* __restrict__ Crow, int lane, int ntile, const float o[8]) {
    const int n  = ntile * 16 + (lane & 15);
    const int hf = lane >> 4;
#pragma unroll
    for (int v = 0; v < 8; ++v) Crow[(v + hf * 8) * ND + n] = o[v];
}

// ---------- prep kernels ----------

__global__ void k_wprep(const float* __restrict__ W, __bf16* __restrict__ Wt) {
    int idx = blockIdx.x * 256 + threadIdx.x;   // 16384 elems
    int k = idx >> 7, n = idx & 127;
    Wt[n * ND + k] = f2bf(W[k * ND + n]);
}

__global__ void k_bias3(const float* __restrict__ a, const float* __restrict__ b,
                        const float* __restrict__ c, float* __restrict__ o) {
    int i = threadIdx.x;
    o[i] = a[i] + b[i] + c[i];
}

__global__ void k_zero(float* __restrict__ p, int n) {
    int i = blockIdx.x * 256 + threadIdx.x;
    if (i < n) p[i] = 0.0f;
}

// ---------- GGNN layer kernels ----------

// t = h @ Wa ; m = t (self-loop init, weight 1). One wave = one 16-row tile.
__global__ void __launch_bounds__(256) k_gemm_t(const float* __restrict__ h, const __bf16* __restrict__ WaT,
                                                float* __restrict__ t, float* __restrict__ m, int ntiles) {
    const int wave = threadIdx.x >> 5, lane = threadIdx.x & 31;
    const int rt = blockIdx.x * 8 + wave;
    if (rt >= ntiles) return;
    const size_t ro = (size_t)rt * 16 * ND;
    const float* Arow = h + ro;
    v16bf a[4];
#pragma unroll
    for (int k = 0; k < 4; ++k) a[k] = load_afrag(Arow, lane, k * 32);
    float* Trow = t + ro;
    float* Mrow = m + ro;
#pragma unroll
    for (int nt = 0; nt < 8; ++nt) {
        v8f acc = {};
#pragma unroll
        for (int k = 0; k < 4; ++k) acc = wmma_bf16(a[k], load_bfrag(WaT, lane, nt, k * 32), acc);
        float o[8];
#pragma unroll
        for (int v = 0; v < 8; ++v) o[v] = acc[v];
        store_ctile(Trow, lane, nt, o);
        store_ctile(Mrow, lane, nt, o);
    }
}

// m[dst] += t[src] * ew  (32 threads per edge, 4 channels each)
__global__ void k_scatter(const float* __restrict__ t, const int* __restrict__ ei,
                          const float* __restrict__ ew, float* __restrict__ m, int E) {
    long long gid = (long long)blockIdx.x * blockDim.x + threadIdx.x;
    int e  = (int)(gid >> 5);
    int c4 = (int)(gid & 31) * 4;
    if (e >= E) return;
    int s = ei[e];
    int d = ei[E + e];
    float w = ew[e];
    float4 tv = *(const float4*)(t + (size_t)s * ND + c4);
    float* mp = m + (size_t)d * ND + c4;
    atomicAdd(mp + 0, tv.x * w);
    atomicAdd(mp + 1, tv.y * w);
    atomicAdd(mp + 2, tv.z * w);
    atomicAdd(mp + 3, tv.w * w);
}

// Fused GRU step (one wave = one 16-row tile; z in VGPRs, rh/ci staged in LDS):
//   z  = sigmoid(m@Wzm + h@Wzs + bz)
//   rh = sigmoid(m@Wrm + h@Wrs + br) * h          -> LDS
//   ci = m@Whm + rh@Whs + bh                      -> LDS
//   h  = z * ((ci@Wl + bl) * sigmoid(ci@Wg + bg)) + (1-z)*h   (in place)
__global__ void __launch_bounds__(32) k_fused(const float* __restrict__ m, float* __restrict__ h,
        const __bf16* __restrict__ Wzm, const __bf16* __restrict__ Wzs,
        const __bf16* __restrict__ Wrm, const __bf16* __restrict__ Wrs,
        const __bf16* __restrict__ Whm, const __bf16* __restrict__ Whs,
        const __bf16* __restrict__ Wl,  const __bf16* __restrict__ Wg,
        const float* __restrict__ bz, const float* __restrict__ br, const float* __restrict__ bh,
        const float* __restrict__ bl, const float* __restrict__ bg, int ntiles) {
    __shared__ __align__(16) float lds_a[16 * ND];   // 8KB staging: rh, then ci
    const int lane = threadIdx.x;
    const int rt = blockIdx.x;
    if (rt >= ntiles) return;   // single-wave WG: barriers are S_NOP, no deadlock
    const size_t ro = (size_t)rt * 16 * ND;
    const float* Mrow = m + ro;
    float* Hrow = h + ro;
    const int hf = lane >> 4;

    v16bf am[4], ah[4];
#pragma unroll
    for (int k = 0; k < 4; ++k) {
        am[k] = load_afrag(Mrow, lane, k * 32);
        ah[k] = load_afrag(Hrow, lane, k * 32);
    }

    float zreg[8][8];

    // ---- pass 1: z (registers) and rh (LDS) ----
#pragma unroll
    for (int nt = 0; nt < 8; ++nt) {
        v8f az = {}, ar = {};
#pragma unroll
        for (int k = 0; k < 4; ++k) {
            az = wmma_bf16(am[k], load_bfrag(Wzm, lane, nt, k * 32), az);
            az = wmma_bf16(ah[k], load_bfrag(Wzs, lane, nt, k * 32), az);
            ar = wmma_bf16(am[k], load_bfrag(Wrm, lane, nt, k * 32), ar);
            ar = wmma_bf16(ah[k], load_bfrag(Wrs, lane, nt, k * 32), ar);
        }
        const int n = nt * 16 + (lane & 15);
        const float bzn = bz[n], brn = br[n];
        float hv[8];
        load_ctile(Hrow, lane, nt, hv);
#pragma unroll
        for (int v = 0; v < 8; ++v) {
            zreg[nt][v] = sigm(az[v] + bzn);
            lds_a[(v + hf * 8) * ND + n] = sigm(ar[v] + brn) * hv[v];
        }
    }
    __syncthreads();

    v16bf arh[4];
#pragma unroll
    for (int k = 0; k < 4; ++k) arh[k] = load_afrag(lds_a, lane, k * 32);
    __syncthreads();   // everyone done reading rh before ci overwrites

    // ---- pass 2: ci -> LDS ----
#pragma unroll
    for (int nt = 0; nt < 8; ++nt) {
        v8f acc = {};
#pragma unroll
        for (int k = 0; k < 4; ++k) {
            acc = wmma_bf16(am[k],  load_bfrag(Whm, lane, nt, k * 32), acc);
            acc = wmma_bf16(arh[k], load_bfrag(Whs, lane, nt, k * 32), acc);
        }
        const int n = nt * 16 + (lane & 15);
        const float bn = bh[n];
#pragma unroll
        for (int v = 0; v < 8; ++v) lds_a[(v + hf * 8) * ND + n] = acc[v] + bn;
    }
    __syncthreads();

    v16bf aci[4];
#pragma unroll
    for (int k = 0; k < 4; ++k) aci[k] = load_afrag(lds_a, lane, k * 32);

    // ---- pass 3: candidate + gate, h update (in place) ----
#pragma unroll
    for (int nt = 0; nt < 8; ++nt) {
        v8f al = {}, ag = {};
#pragma unroll
        for (int k = 0; k < 4; ++k) {
            al = wmma_bf16(aci[k], load_bfrag(Wl, lane, nt, k * 32), al);
            ag = wmma_bf16(aci[k], load_bfrag(Wg, lane, nt, k * 32), ag);
        }
        const int n = nt * 16 + (lane & 15);
        const float bln = bl[n], bgn = bg[n];
        float hv[8], o[8];
        load_ctile(Hrow, lane, nt, hv);
#pragma unroll
        for (int v = 0; v < 8; ++v) {
            const float ht = (al[v] + bln) * sigm(ag[v] + bgn);
            const float zv = zreg[nt][v];
            o[v] = zv * ht + (1.0f - zv) * hv[v];
        }
        store_ctile(Hrow, lane, nt, o);
    }
}

// ---------- head ----------

// ha = h * sigmoid(h . Wg + bg)   (one block per node)
__global__ void k_att(const float* __restrict__ h, const float* __restrict__ Wg,
                      const float* __restrict__ bg, float* __restrict__ ha) {
    __shared__ float red[128];
    int n = blockIdx.x, c = threadIdx.x;
    float hv = h[(size_t)n * ND + c];
    red[c] = hv * Wg[c];
    __syncthreads();
    for (int s = 64; s > 0; s >>= 1) {
        if (c < s) red[c] += red[c + s];
        __syncthreads();
    }
    float att = sigm(red[0] + bg[0]);
    ha[(size_t)n * ND + c] = hv * att;
}

// e = relu(ha @ We + be)
__global__ void __launch_bounds__(256) k_attemb(const float* __restrict__ ha, const __bf16* __restrict__ We,
        const float* __restrict__ be, float* __restrict__ e, int ntiles) {
    const int wave = threadIdx.x >> 5, lane = threadIdx.x & 31;
    const int rt = blockIdx.x * 8 + wave;
    if (rt >= ntiles) return;
    const size_t ro = (size_t)rt * 16 * ND;
    const float* Arow = ha + ro;
    v16bf a[4];
#pragma unroll
    for (int k = 0; k < 4; ++k) a[k] = load_afrag(Arow, lane, k * 32);
    float* Erow = e + ro;
#pragma unroll
    for (int nt = 0; nt < 8; ++nt) {
        v8f acc = {};
#pragma unroll
        for (int k = 0; k < 4; ++k) acc = wmma_bf16(a[k], load_bfrag(We, lane, nt, k * 32), acc);
        const int n = nt * 16 + (lane & 15);
        const float bn = be[n];
        float o[8];
#pragma unroll
        for (int v = 0; v < 8; ++v) o[v] = fmaxf(acc[v] + bn, 0.0f);
        store_ctile(Erow, lane, nt, o);
    }
}

// segment max / sum / count (e >= 0, max-init 0 => int atomicMax trick valid)
__global__ void k_pool(const float* __restrict__ e, const int* __restrict__ batch,
                       float* __restrict__ xmax, float* __restrict__ xsum,
                       float* __restrict__ cnt, int N) {
    long long gid = (long long)blockIdx.x * blockDim.x + threadIdx.x;
    int n  = (int)(gid >> 5);
    int c4 = (int)(gid & 31) * 4;
    if (n >= N) return;
    int g = batch[n];
    float4 v = *(const float4*)(e + (size_t)n * ND + c4);
    int* mx = (int*)(xmax + (size_t)g * ND + c4);
    atomicMax(mx + 0, __float_as_int(v.x));
    atomicMax(mx + 1, __float_as_int(v.y));
    atomicMax(mx + 2, __float_as_int(v.z));
    atomicMax(mx + 3, __float_as_int(v.w));
    float* sp = xsum + (size_t)g * ND + c4;
    atomicAdd(sp + 0, v.x);
    atomicAdd(sp + 1, v.y);
    atomicAdd(sp + 2, v.z);
    atomicAdd(sp + 3, v.w);
    if (c4 == 0) atomicAdd(cnt + g, 1.0f);
}

// out = relu([max, mean] @ W1 + b1) @ W2 + b2   (one block per graph)
__global__ void k_mlp(const float* __restrict__ xmax, const float* __restrict__ xsum,
                      const float* __restrict__ cnt,
                      const float* __restrict__ W1, const float* __restrict__ b1,
                      const float* __restrict__ W2, const float* __restrict__ b2,
                      float* __restrict__ out) {
    __shared__ float pool[256];
    __shared__ float hid[128];
    int g = blockIdx.x, j = threadIdx.x;   // 128 threads
    float inv = 1.0f / fmaxf(cnt[g], 1.0f);
    pool[j]       = fmaxf(xmax[g * ND + j], 0.0f);
    pool[128 + j] = xsum[g * ND + j] * inv;
    __syncthreads();
    float a = b1[j];
    for (int k = 0; k < 256; ++k) a += pool[k] * W1[k * ND + j];
    hid[j] = fmaxf(a, 0.0f);
    __syncthreads();
    if (j < 16) {
        float o = b2[j];
        for (int k = 0; k < 128; ++k) o += hid[k] * W2[k * 16 + j];
        out[g * 16 + j] = o;
    }
}

// ---------- host ----------

extern "C" void kernel_launch(void* const* d_in, const int* in_sizes, int n_in,
                              void* d_out, int out_size, void* d_ws, size_t ws_size,
                              hipStream_t stream) {
    const float* x    = (const float*)d_in[0];
    const int*   ei   = (const int*)  d_in[1];
    const int*   bat  = (const int*)  d_in[2];
    const float* ew   = (const float*)d_in[3];
    const float* Wa   = (const float*)d_in[4];
    const float* Wzm  = (const float*)d_in[5];  const float* bzm = (const float*)d_in[6];
    const float* Wzs  = (const float*)d_in[7];  const float* bzs = (const float*)d_in[8];
    const float* Wrm  = (const float*)d_in[9];  const float* brm = (const float*)d_in[10];
    const float* Wrs  = (const float*)d_in[11]; const float* brs = (const float*)d_in[12];
    const float* Whm  = (const float*)d_in[13]; const float* bhm = (const float*)d_in[14];
    const float* Whs  = (const float*)d_in[15]; const float* bhs = (const float*)d_in[16];
    const float* Whg  = (const float*)d_in[17]; const float* bhg = (const float*)d_in[18];
    const float* Whl  = (const float*)d_in[19]; const float* bhl = (const float*)d_in[20];
    const float* bias_z = (const float*)d_in[21];
    const float* bias_r = (const float*)d_in[22];
    const float* bias_h = (const float*)d_in[23];
    const float* Wag  = (const float*)d_in[24]; const float* bag = (const float*)d_in[25];
    const float* Wae  = (const float*)d_in[26]; const float* bae = (const float*)d_in[27];
    const float* W1   = (const float*)d_in[28]; const float* b1  = (const float*)d_in[29];
    const float* W2   = (const float*)d_in[30]; const float* b2  = (const float*)d_in[31];
    float* out = (float*)d_out;

    const int N = in_sizes[0] / ND;   // 50000
    const int E = in_sizes[3];        // 500000
    const int G = 64;
    const int ntiles = N / 16;        // N is a multiple of 16

    // workspace carve-up
    char* ws = (char*)d_ws;
    size_t off = 0;
    auto carve = [&](size_t bytes) -> void* {
        void* p = ws + off;
        off = (off + bytes + 255) & ~(size_t)255;
        return p;
    };
    const size_t wtb = (size_t)ND * ND * sizeof(__bf16);
    __bf16* WaT  = (__bf16*)carve(wtb);
    __bf16* WzmT = (__bf16*)carve(wtb);
    __bf16* WzsT = (__bf16*)carve(wtb);
    __bf16* WrmT = (__bf16*)carve(wtb);
    __bf16* WrsT = (__bf16*)carve(wtb);
    __bf16* WhmT = (__bf16*)carve(wtb);
    __bf16* WhsT = (__bf16*)carve(wtb);
    __bf16* WhgT = (__bf16*)carve(wtb);
    __bf16* WhlT = (__bf16*)carve(wtb);
    __bf16* WeT  = (__bf16*)carve(wtb);
    float* cbz = (float*)carve(ND * sizeof(float));
    float* cbr = (float*)carve(ND * sizeof(float));
    float* cbh = (float*)carve(ND * sizeof(float));
    const size_t nodeb = (size_t)N * ND * sizeof(float);
    float* hbuf  = (float*)carve(nodeb);
    float* tbuf  = (float*)carve(nodeb);   // also reused as ha
    float* mbuf  = (float*)carve(nodeb);   // also reused as e
    float* xmax  = (float*)carve((size_t)G * ND * sizeof(float));
    float* xsum  = (float*)carve((size_t)G * ND * sizeof(float));
    float* cnt   = (float*)carve((size_t)G * sizeof(float));

    // --- prep: bf16 transposed weights + combined biases, h = x ---
    const float* wsrc[10] = {Wa, Wzm, Wzs, Wrm, Wrs, Whm, Whs, Whg, Whl, Wae};
    __bf16*      wdst[10] = {WaT, WzmT, WzsT, WrmT, WrsT, WhmT, WhsT, WhgT, WhlT, WeT};
    for (int i = 0; i < 10; ++i)
        k_wprep<<<64, 256, 0, stream>>>(wsrc[i], wdst[i]);
    k_bias3<<<1, 128, 0, stream>>>(bzm, bzs, bias_z, cbz);
    k_bias3<<<1, 128, 0, stream>>>(brm, brs, bias_r, cbr);
    k_bias3<<<1, 128, 0, stream>>>(bhm, bhs, bias_h, cbh);
    hipMemcpyAsync(hbuf, x, nodeb, hipMemcpyDeviceToDevice, stream);

    const int gblocks = (ntiles + 7) / 8;   // 8 waves (16-row tiles) per 256-thread block
    const long long ework = (long long)E * 32;
    const int eblocks = (int)((ework + 255) / 256);

    for (int L = 0; L < 3; ++L) {
        k_gemm_t<<<gblocks, 256, 0, stream>>>(hbuf, WaT, tbuf, mbuf, ntiles);
        k_scatter<<<eblocks, 256, 0, stream>>>(tbuf, ei, ew, mbuf, E);
        k_fused<<<ntiles, 32, 0, stream>>>(mbuf, hbuf,
                                           WzmT, WzsT, WrmT, WrsT, WhmT, WhsT, WhlT, WhgT,
                                           cbz, cbr, cbh, bhl, bhg, ntiles);
    }

    // head
    k_att<<<N, 128, 0, stream>>>(hbuf, Wag, bag, tbuf);                   // ha in tbuf
    k_attemb<<<gblocks, 256, 0, stream>>>(tbuf, WeT, bae, mbuf, ntiles);  // e in mbuf
    k_zero<<<(G * ND + 255) / 256, 256, 0, stream>>>(xmax, G * ND);
    k_zero<<<(G * ND + 255) / 256, 256, 0, stream>>>(xsum, G * ND);
    k_zero<<<1, 256, 0, stream>>>(cnt, G);
    {
        const long long pwork = (long long)N * 32;
        const int pblocks = (int)((pwork + 255) / 256);
        k_pool<<<pblocks, 256, 0, stream>>>(mbuf, bat, xmax, xsum, cnt, N);
    }
    k_mlp<<<G, 128, 0, stream>>>(xmax, xsum, cnt, W1, b1, W2, b2, out);
    (void)n_in; (void)out_size; (void)ws_size;
}
